// MixtralSparseMoe_46883863003222
// MI455X (gfx1250) — matrix operand order repaired and checked by reading, not compiled, not run on hardware
//
#include <hip/hip_runtime.h>
#include <hip/hip_bf16.h>

// ---------------- problem constants (from reference) ----------------
constexpr int kE = 8;      // experts
constexpr int kH = 1024;   // hidden
constexpr int kF = 3584;   // expert intermediate
constexpr int kT = 1024;   // tokens = B*S

constexpr int kChunk   = 256;          // K-chunk staged through LDS
constexpr int kAStride = kChunk + 8;   // padded row stride (elements) -> no bank conflicts

typedef __attribute__((ext_vector_type(16))) __bf16 v16bf;
typedef __attribute__((ext_vector_type(8)))  __bf16 v8bf;
typedef __attribute__((ext_vector_type(8)))  float  v8f;

// -------- helpers: build v16bf lane operands for wmma_f32_16x16x32_bf16 -----
__device__ __forceinline__ v16bf cvt16(float4 a0, float4 a1, float4 b0, float4 b1) {
  v16bf r;
  r[0]  = (__bf16)a0.x; r[1]  = (__bf16)a0.y; r[2]  = (__bf16)a0.z; r[3]  = (__bf16)a0.w;
  r[4]  = (__bf16)a1.x; r[5]  = (__bf16)a1.y; r[6]  = (__bf16)a1.z; r[7]  = (__bf16)a1.w;
  r[8]  = (__bf16)b0.x; r[9]  = (__bf16)b0.y; r[10] = (__bf16)b0.z; r[11] = (__bf16)b0.w;
  r[12] = (__bf16)b1.x; r[13] = (__bf16)b1.y; r[14] = (__bf16)b1.z; r[15] = (__bf16)b1.w;
  return r;
}

// B-matrix (16-bit, 32x16): lane = column n; elems 0..7 -> K = half*8 + 0..7,
// elems 8..15 -> K = 16 + half*8 + 0..7.
__device__ __forceinline__ v16bf loadB_f32(const float* __restrict__ row, int k0, int half16) {
  const float4* p = (const float4*)(row + k0 + half16 / 2);
  const float4* q = (const float4*)(row + k0 + 16 + half16 / 2);
  return cvt16(p[0], p[1], q[0], q[1]);
}

// A-matrix (16-bit, 16x32) from LDS bf16: elems 0..7 -> K = half*8 + 0..7,
// elems 8..15 -> K = 16 + half*8 + 0..7.  (two ds_load_b128 per fragment)
__device__ __forceinline__ v16bf loadA_lds(const __bf16* row, int kk, int half8) {
  v8bf lo = *(const v8bf*)(row + kk + half8);
  v8bf hi = *(const v8bf*)(row + kk + 16 + half8);
  v16bf r;
#pragma unroll
  for (int i = 0; i < 8; ++i) { r[i] = lo[i]; r[i + 8] = hi[i]; }
  return r;
}

// ---- CDNA5 async global->LDS copy (ASYNCcnt-tracked DMA, bypasses VGPRs) ----
__device__ __forceinline__ void async_copy_b128(const void* gaddr, void* lds) {
  // VDST = per-lane LDS byte offset (low 32 bits of the generic shared pointer)
  unsigned int ldsOff = (unsigned int)(unsigned long long)lds;
  asm volatile("global_load_async_to_lds_b128 %0, %1, off"
               :
               : "v"(ldsOff), "v"(gaddr)
               : "memory");
}
__device__ __forceinline__ void wait_async_le4() {
  asm volatile("s_wait_asynccnt 0x4" ::: "memory");
}
__device__ __forceinline__ void wait_async_0() {
  asm volatile("s_wait_asynccnt 0x0" ::: "memory");
}

// ---------------- pass 1: router (logits, softmax, top-2, expert lists) ------
__global__ void router_gate_kernel(const float* __restrict__ x,
                                   const float* __restrict__ gate_w,
                                   float* __restrict__ logits_out,
                                   int* __restrict__ counts,
                                   int* __restrict__ list,
                                   float* __restrict__ wlist) {
  int t = blockIdx.x * blockDim.x + threadIdx.x;
  if (t >= kT) return;
  const float* xr = x + (size_t)t * kH;

  float acc[kE];
#pragma unroll
  for (int e = 0; e < kE; ++e) acc[e] = 0.f;
  for (int h = 0; h < kH; h += 4) {
    float4 xv = *(const float4*)(xr + h);
#pragma unroll
    for (int e = 0; e < kE; ++e) {
      float4 g = *(const float4*)(gate_w + (size_t)e * kH + h);
      acc[e] += xv.x * g.x + xv.y * g.y + xv.z * g.z + xv.w * g.w;
    }
  }
#pragma unroll
  for (int e = 0; e < kE; ++e) logits_out[(size_t)t * kE + e] = acc[e];

  float mx = acc[0];
#pragma unroll
  for (int e = 1; e < kE; ++e) mx = fmaxf(mx, acc[e]);
  float p[kE];
#pragma unroll
  for (int e = 0; e < kE; ++e) p[e] = __expf(acc[e] - mx);

  int i0 = 0;
#pragma unroll
  for (int e = 1; e < kE; ++e) if (p[e] > p[i0]) i0 = e;
  int i1 = (i0 == 0) ? 1 : 0;
#pragma unroll
  for (int e = 0; e < kE; ++e) if (e != i0 && p[e] > p[i1]) i1 = e;

  float denom = p[i0] + p[i1];
  float w0 = p[i0] / denom;
  float w1 = p[i1] / denom;

  int s0 = atomicAdd(&counts[i0], 1);
  list[i0 * kT + s0]  = t * 2 + 0;
  wlist[i0 * kT + s0] = w0;
  int s1 = atomicAdd(&counts[i1], 1);
  list[i1 * kT + s1]  = t * 2 + 1;
  wlist[i1 * kT + s1] = w1;
}

// ------- pass 2: h = silu(x w1^T) * (x w3^T) for gathered tokens, bf16 out ---
// grid: (kF/256, kT/16, kE); block: 128 (4 waves; wave w -> 16 tokens x 64 F)
__global__ void __launch_bounds__(128)
moe_up_swiglu_kernel(const float* __restrict__ x,
                     const float* __restrict__ w1,
                     const float* __restrict__ w3,
                     const int* __restrict__ counts,
                     const int* __restrict__ list,
                     __bf16* __restrict__ hbuf) {
  __shared__ __attribute__((aligned(16))) __bf16 sA[16 * kAStride];

  const int e    = blockIdx.z;
  const int cnt  = counts[e];
  const int row0 = blockIdx.y * 16;
  if (row0 >= cnt) return;                  // block-uniform exit (before barriers)

  const int tid    = threadIdx.x;
  const int wave   = tid >> 5;
  const int lane   = tid & 31;
  const int idx    = lane & 15;
  const int half8  = (lane >> 4) * 8;
  const int half16 = (lane >> 4) * 16;
  const int fw     = blockIdx.x * 256 + wave * 64;

  const int* elist = list + e * kT;

  // staging assignment: 8 threads per row, 32 K-elems per thread
  const int srow = tid >> 3;
  const int scol = (tid & 7) * 32;
  int slotS = row0 + srow;
  int pairS = elist[(slotS < cnt) ? slotS : 0];  // clamp: garbage rows never stored
  const float* xs = x + (size_t)(pairS >> 1) * kH;

  const float* w1e = w1 + (size_t)e * kF * kH;
  const float* w3e = w3 + (size_t)e * kF * kH;
  const float* b1p[4];
  const float* b3p[4];
#pragma unroll
  for (int nf = 0; nf < 4; ++nf) {
    int f = fw + nf * 16 + idx;
    b1p[nf] = w1e + (size_t)f * kH;
    b3p[nf] = w3e + (size_t)f * kH;
  }

  v8f acc1[4] = {};
  v8f acc3[4] = {};
  const __bf16* arow = &sA[idx * kAStride];

  for (int c0 = 0; c0 < kH; c0 += kChunk) {
    // cooperative stage: convert 16 x kChunk f32 -> bf16 into LDS
#pragma unroll
    for (int u = 0; u < 32; u += 4) {
      float4 v = *(const float4*)(xs + c0 + scol + u);
      __bf16* d = &sA[srow * kAStride + scol + u];
      d[0] = (__bf16)v.x; d[1] = (__bf16)v.y; d[2] = (__bf16)v.z; d[3] = (__bf16)v.w;
    }
    __syncthreads();
#pragma unroll 2
    for (int kk = 0; kk < kChunk; kk += 32) {
      const int k0 = c0 + kk;
      v16bf a = loadA_lds(arow, kk, half8);
#pragma unroll
      for (int nf = 0; nf < 4; ++nf) {
        __builtin_prefetch(b1p[nf] + k0 + 64, 0, 1);
        v16bf b1 = loadB_f32(b1p[nf], k0, half16);
        acc1[nf] = __builtin_amdgcn_wmma_f32_16x16x32_bf16(
            false, a, false, b1, (short)0, acc1[nf], false, false);
        __builtin_prefetch(b3p[nf] + k0 + 64, 0, 1);
        v16bf b3 = loadB_f32(b3p[nf], k0, half16);
        acc3[nf] = __builtin_amdgcn_wmma_f32_16x16x32_bf16(
            false, a, false, b3, (short)0, acc3[nf], false, false);
      }
    }
    __syncthreads();
  }

  // C/D layout: elem j -> (m = j + half8, n = idx)
  int prow[8];
#pragma unroll
  for (int j = 0; j < 8; ++j) {
    int slot = row0 + half8 + j;
    prow[j] = (slot < cnt) ? elist[slot] : -1;
  }
#pragma unroll
  for (int nf = 0; nf < 4; ++nf) {
    int f = fw + nf * 16 + idx;
#pragma unroll
    for (int j = 0; j < 8; ++j) {
      if (prow[j] >= 0) {
        float g = acc1[nf][j];
        float u = acc3[nf][j];
        float hv = (g / (1.f + __expf(-g))) * u;   // silu(g) * u
        hbuf[(size_t)prow[j] * kF + f] = (__bf16)hv;
      }
    }
  }
}

// ------- pass 3: y += weight * (h w2^T), atomic combine into output ----------
// grid: (kH/256, kT/16, kE); block: 128 (4 waves; wave w -> 16 tokens x 64 H)
// A-tile (bf16 h rows) staged via double-buffered GLOBAL_LOAD_ASYNC_TO_LDS.
__global__ void __launch_bounds__(128)
moe_down_combine_kernel(const __bf16* __restrict__ hbuf,
                        const float* __restrict__ w2,
                        const int* __restrict__ counts,
                        const int* __restrict__ list,
                        const float* __restrict__ wlist,
                        float* __restrict__ out) {
  __shared__ __attribute__((aligned(16))) __bf16 sA[2][16 * kAStride];

  const int e    = blockIdx.z;
  const int cnt  = counts[e];
  const int row0 = blockIdx.y * 16;
  if (row0 >= cnt) return;

  const int tid    = threadIdx.x;
  const int wave   = tid >> 5;
  const int lane   = tid & 31;
  const int idx    = lane & 15;
  const int half8  = (lane >> 4) * 8;
  const int half16 = (lane >> 4) * 16;
  const int hw     = blockIdx.x * 256 + wave * 64;

  const int*   elist = list  + e * kT;
  const float* ewl   = wlist + e * kT;

  const int srow = tid >> 3;
  const int scol = (tid & 7) * 32;
  int slotS = row0 + srow;
  int pairS = elist[(slotS < cnt) ? slotS : 0];
  const __bf16* hs = hbuf + (size_t)pairS * kF;

  const float* w2e = w2 + (size_t)e * kH * kF;
  const float* bp[4];
#pragma unroll
  for (int nh = 0; nh < 4; ++nh)
    bp[nh] = w2e + (size_t)(hw + nh * 16 + idx) * kF;

  v8f acc[4] = {};

  constexpr int nChunks = kF / kChunk;   // 14

  // prime chunk 0 into buffer 0 (4 async b128 per thread -> ASYNCcnt=4/wave)
#pragma unroll
  for (int u = 0; u < 32; u += 8)
    async_copy_b128(hs + scol + u, &sA[0][srow * kAStride + scol + u]);

  for (int c = 0; c < nChunks; ++c) {
    const int cur = c & 1;
    if (c + 1 < nChunks) {
      // kick off next chunk into the other buffer, then wait for current only
      const int cn = (c + 1) * kChunk;
#pragma unroll
      for (int u = 0; u < 32; u += 8)
        async_copy_b128(hs + cn + scol + u, &sA[1 - cur][srow * kAStride + scol + u]);
      wait_async_le4();    // in-order completion: <=4 outstanding => chunk c done
    } else {
      wait_async_0();
    }
    __syncthreads();       // each wave waited its own ASYNCcnt -> all data visible

    const __bf16* arow = &sA[cur][idx * kAStride];
    const int c0 = c * kChunk;
#pragma unroll 2
    for (int kk = 0; kk < kChunk; kk += 32) {
      const int k0 = c0 + kk;
      v16bf a = loadA_lds(arow, kk, half8);
#pragma unroll
      for (int nh = 0; nh < 4; ++nh) {
        __builtin_prefetch(bp[nh] + k0 + 64, 0, 1);
        v16bf b = loadB_f32(bp[nh], k0, half16);
        acc[nh] = __builtin_amdgcn_wmma_f32_16x16x32_bf16(
            false, a, false, b, (short)0, acc[nh], false, false);
      }
    }
    __syncthreads();       // all waves done reading sA[cur] before it is refilled
  }

  int   prow[8];
  float wrow[8];
#pragma unroll
  for (int j = 0; j < 8; ++j) {
    int slot = row0 + half8 + j;
    bool v = slot < cnt;
    prow[j] = v ? elist[slot] : -1;
    wrow[j] = v ? ewl[slot] : 0.f;
  }
#pragma unroll
  for (int nh = 0; nh < 4; ++nh) {
    int hcol = hw + nh * 16 + idx;
#pragma unroll
    for (int j = 0; j < 8; ++j) {
      if (prow[j] >= 0) {
        int t = prow[j] >> 1;
        atomicAdd(&out[(size_t)t * kH + hcol], acc[nh][j] * wrow[j]);
      }
    }
  }
}

// ---------------------------------------------------------------------------
extern "C" void kernel_launch(void* const* d_in, const int* in_sizes, int n_in,
                              void* d_out, int out_size, void* d_ws, size_t ws_size,
                              hipStream_t stream) {
  (void)in_sizes; (void)n_in; (void)out_size; (void)ws_size;
  const float* x      = (const float*)d_in[0];  // [T, H]
  const float* gate_w = (const float*)d_in[1];  // [E, H]
  const float* w1     = (const float*)d_in[2];  // [E, F, H]
  const float* w2     = (const float*)d_in[3];  // [E, H, F]
  const float* w3     = (const float*)d_in[4];  // [E, F, H]

  float* out    = (float*)d_out;                // [T*H] final ++ [T*E] logits
  float* logits = out + (size_t)kT * kH;

  // workspace layout
  char*   ws     = (char*)d_ws;
  int*    counts = (int*)ws;                                      // E ints
  int*    list   = (int*)(ws + 256);                              // E*T ints
  float*  wlist  = (float*)(ws + 256 + (size_t)kE * kT * 4);      // E*T floats
  __bf16* hbuf   = (__bf16*)(ws + 256 + 2 * (size_t)kE * kT * 4); // 2T x F bf16

  hipMemsetAsync(counts, 0, kE * sizeof(int), stream);
  hipMemsetAsync(out, 0, (size_t)kT * kH * sizeof(float), stream);

  router_gate_kernel<<<(kT + 127) / 128, 128, 0, stream>>>(
      x, gate_w, logits, counts, list, wlist);

  dim3 gUp(kF / 256, kT / 16, kE);   // 14 x 64 x 8
  moe_up_swiglu_kernel<<<gUp, 128, 0, stream>>>(x, w1, w3, counts, list, hbuf);

  dim3 gDn(kH / 256, kT / 16, kE);   // 4 x 64 x 8
  moe_down_combine_kernel<<<gDn, 128, 0, stream>>>(hbuf, w2, counts, list, wlist, out);
}
